// CausalSelfAttention_21079699489189
// MI455X (gfx1250) — compile-verified, matrix operand order checked by reading, not run on hardware
//
#include <hip/hip_runtime.h>

// ---------------------------------------------------------------------------
// Types
// ---------------------------------------------------------------------------
typedef __bf16 bf16;
typedef __attribute__((ext_vector_type(8)))  bf16  v8bf;
typedef __attribute__((ext_vector_type(16))) bf16  v16bf;
typedef __attribute__((ext_vector_type(8)))  float v8f;
typedef __attribute__((ext_vector_type(4)))  unsigned int u32x4;
typedef __attribute__((ext_vector_type(8)))  int i32x8;
typedef __attribute__((ext_vector_type(4)))  int i32x4;

#if __has_builtin(__builtin_amdgcn_tensor_load_to_lds) && \
    __has_builtin(__builtin_amdgcn_s_wait_tensorcnt)
#define USE_TDM 1
#else
#define USE_TDM 0
#endif

__device__ __forceinline__ bf16 f2bf(float f) {
  union { float f; unsigned u; } v; v.f = f;
  unsigned r = v.u + 0x7FFFu + ((v.u >> 16) & 1u);   // round-to-nearest-even
  unsigned short h = (unsigned short)(r >> 16);
  return __builtin_bit_cast(bf16, h);
}

__device__ __forceinline__ v8f wmma_bf16(v16bf a, v16bf b, v8f c) {
  return __builtin_amdgcn_wmma_f32_16x16x32_bf16(
      false, a, false, b, (short)0, c, false, false);
}

// 16-bit A/B fragment (16 rows x K=32): lane L holds row (L&15); K chunks
// [8*(L>>4), +8) and [16+8*(L>>4), +8) -> two b128 loads + shuffle (no moves).
__device__ __forceinline__ v16bf load_frag(const bf16* rowp, int laneHi) {
  const bf16* p = rowp + laneHi * 8;
  v8bf lo = *reinterpret_cast<const v8bf*>(p);
  v8bf hi = *reinterpret_cast<const v8bf*>(p + 16);
  return __builtin_shufflevector(lo, hi, 0, 1, 2, 3, 4, 5, 6, 7,
                                 8, 9, 10, 11, 12, 13, 14, 15);
}

#if USE_TDM
// ---------------------------------------------------------------------------
// Tensor Data Mover: 2D tile (bf16) global -> LDS.
// D# bit packing per CDNA5 ISA §8.3 (group0) / §8.4 (group1).
// pad codes: interval 3 -> pad every 16 DWORDs; amount 3 -> insert 4 DWORDs.
// This toolchain's builtin is the 6-arg form:
//   (u32x4 g0, i32x8 g1, i32x4 g2, i32x4 g3, i32x8 pad, i32 cpol)
// ---------------------------------------------------------------------------
__device__ __forceinline__ void tdm_load_2d_bf16(
    unsigned lds_off, const bf16* gptr,
    unsigned tile_d0, unsigned tile_d1,          // elements (x, y)
    unsigned long long stride0_elems,            // row stride (elements)
    bool pad_en, unsigned pad_interval_code, unsigned pad_amount_code) {
  unsigned long long ga = (unsigned long long)(uintptr_t)gptr;
  u32x4 g0;
  g0[0] = 1u;                                        // count=1, user D#
  g0[1] = lds_off;                                   // lds_addr (bytes)
  g0[2] = (unsigned)(ga & 0xffffffffu);              // global_addr[31:0]
  g0[3] = (unsigned)((ga >> 32) & 0x01ffffffu)       // global_addr[56:32]
        | (2u << 30);                                // type = 2 ("image")
  const unsigned td0 = tile_d0, td1 = tile_d1;       // tensor dims = tile dims
  unsigned w0 = (1u << 16)                           // data_size = 1 -> 2 bytes
              | (pad_en ? (1u << 20) : 0u)
              | (pad_interval_code << 22)
              | (pad_amount_code << 25);
  i32x8 g1;
  g1[0] = (int)w0;
  g1[1] = (int)((td0 & 0xffffu) << 16);                            // dim0 lo16
  g1[2] = (int)(((td0 >> 16) & 0xffffu) | ((td1 & 0xffffu) << 16));// dim0 hi/dim1 lo
  g1[3] = (int)(((td1 >> 16) & 0xffffu) | ((tile_d0 & 0xffffu) << 16)); // tile_dim0
  g1[4] = (int)(tile_d1 & 0xffffu);                                // tile_dim1
  g1[5] = (int)(unsigned)(stride0_elems & 0xffffffffu);            // stride0 lo32
  g1[6] = (int)(unsigned)((stride0_elems >> 32) & 0xffffu);        // stride0 hi16
  g1[7] = 0;
  i32x4 z4 = {0, 0, 0, 0};
  i32x8 z8 = {0, 0, 0, 0, 0, 0, 0, 0};
  __builtin_amdgcn_tensor_load_to_lds(g0, g1, z4, z4, z8, 0);
}
#endif

// ---------------------------------------------------------------------------
// fp32 -> bf16 conversion (8 elements / thread)
// ---------------------------------------------------------------------------
__global__ __launch_bounds__(256) void f32_to_bf16_vec(
    const float* __restrict__ in, bf16* __restrict__ out, long long n8) {
  long long idx = (long long)blockIdx.x * blockDim.x + threadIdx.x;
  if (idx >= n8) return;
  const float4* p = reinterpret_cast<const float4*>(in) + idx * 2;
  float4 f0 = p[0], f1 = p[1];
  v8bf o;
  o[0] = f2bf(f0.x); o[1] = f2bf(f0.y); o[2] = f2bf(f0.z); o[3] = f2bf(f0.w);
  o[4] = f2bf(f1.x); o[5] = f2bf(f1.y); o[6] = f2bf(f1.z); o[7] = f2bf(f1.w);
  reinterpret_cast<v8bf*>(out)[idx] = o;
}

// ---------------------------------------------------------------------------
// bf16 GEMM: C[M,N] = A[M,K] * B[K,N], f32 accumulate.
// Block tile 128x128, BK=32, 8 waves (2M x 4N), wave tile 64x32.
// A tile: TDM -> LDS (TDM pad fields give the 40-elem row stride).
// B tile: coalesced b128 loads + transposed LDS store.
// ---------------------------------------------------------------------------
template <bool OUT_BF16>
__global__ __launch_bounds__(256) void gemm_bf16_wmma(
    const bf16* __restrict__ A, const bf16* __restrict__ B,
    void* __restrict__ Cout, int M, int N, int K) {
  constexpr int LS = 40;                 // 80B row stride, 16B aligned
  __shared__ bf16 As[128 * LS];          // [m][k]
  __shared__ bf16 Bs[128 * LS];          // [n][k]  (transposed on store)
  const int tid = threadIdx.x;
  const int wave = tid >> 5, lane = tid & 31;
  const int laneHi = lane >> 4, lane16 = lane & 15;
  const int bm = blockIdx.y * 128, bn = blockIdx.x * 128;
  const int wm = (wave >> 2) * 64, wn = (wave & 3) * 32;

  v8f acc[4][2] = {};

  const int am = tid >> 1, ak = (tid & 1) * 16;   // A fallback: 128 rows x 32 k
  const int bk = tid >> 3, bn0 = (tid & 7) * 16;  // B: 32 k x 128 n

#if USE_TDM
  const unsigned as_off = (unsigned)(uintptr_t)&As[0];
#endif

  for (int k0 = 0; k0 < K; k0 += 32) {
    __syncthreads();                       // previous tile fully consumed
#if USE_TDM
    if (wave == 0)                         // one wave drives the A-tile DMA
      tdm_load_2d_bf16(as_off, A + (size_t)bm * K + k0,
                       /*tile x,y=*/32, 128, /*stride=*/(unsigned long long)K,
                       /*pad=*/true, /*every 16 dwords*/ 3, /*+4 dwords*/ 3);
#else
    const bf16* ga = A + (size_t)(bm + am) * K + k0 + ak;
    v8bf a0 = *reinterpret_cast<const v8bf*>(ga);
    v8bf a1 = *reinterpret_cast<const v8bf*>(ga + 8);
    *reinterpret_cast<v8bf*>(&As[am * LS + ak]) = a0;
    *reinterpret_cast<v8bf*>(&As[am * LS + ak + 8]) = a1;
#endif
    // B tile: load coalesced, store transposed Bs[n][k] = B[k][n]
    const bf16* gb = B + (size_t)(k0 + bk) * N + bn + bn0;
    if (k0 + 32 < K)
      __builtin_prefetch(gb + (size_t)32 * N, 0, 3);   // next tile -> global_prefetch
    v8bf b0 = *reinterpret_cast<const v8bf*>(gb);
    v8bf b1 = *reinterpret_cast<const v8bf*>(gb + 8);
#pragma unroll
    for (int i = 0; i < 8; ++i) Bs[(bn0 + i) * LS + bk] = b0[i];
#pragma unroll
    for (int i = 0; i < 8; ++i) Bs[(bn0 + 8 + i) * LS + bk] = b1[i];
#if USE_TDM
    if (wave == 0) __builtin_amdgcn_s_wait_tensorcnt(0);
    asm volatile("" ::: "memory");
#endif
    __syncthreads();                       // A (TDM) + B (ds) visible to all

    v16bf af[4], bfr[2];
#pragma unroll
    for (int im = 0; im < 4; ++im)
      af[im] = load_frag(&As[(wm + im * 16 + lane16) * LS], laneHi);
#pragma unroll
    for (int in = 0; in < 2; ++in)
      bfr[in] = load_frag(&Bs[(wn + in * 16 + lane16) * LS], laneHi);
#pragma unroll
    for (int im = 0; im < 4; ++im)
#pragma unroll
      for (int in = 0; in < 2; ++in)
        acc[im][in] = wmma_bf16(af[im], bfr[in], acc[im][in]);
  }

  // C layout: VGPR j holds row (laneHi*8 + j), col = lane16
#pragma unroll
  for (int im = 0; im < 4; ++im)
#pragma unroll
    for (int in = 0; in < 2; ++in) {
      const int row0 = bm + wm + im * 16 + laneHi * 8;
      const int col = bn + wn + in * 16 + lane16;
#pragma unroll
      for (int j = 0; j < 8; ++j) {
        if (OUT_BF16)
          reinterpret_cast<bf16*>(Cout)[(size_t)(row0 + j) * N + col] =
              f2bf(acc[im][in][j]);
        else
          reinterpret_cast<float*>(Cout)[(size_t)(row0 + j) * N + col] =
              acc[im][in][j];
      }
    }
}

// ---------------------------------------------------------------------------
// Flash attention (causal). qkv bf16 [B*T, 3C]; y bf16 [B*T, C].
// grid = (T/128, H, B); 8 waves; wave owns 16 query rows, loops 32-key blocks.
// Per-wave TDM stages the V tile (32 keys x 64 d, row stride 3C) into LDS
// while S = Q K^T runs; the wave then waits its own TENSORcnt (no barrier).
// ---------------------------------------------------------------------------
__global__ __launch_bounds__(256) void flash_attn_wmma(
    const bf16* __restrict__ qkv, bf16* __restrict__ y) {
  constexpr int T = 2048, C = 1024, C3 = 3072, D = 64;
  const int tid = threadIdx.x, wave = tid >> 5, lane = tid & 31;
  const int laneHi = lane >> 4, lane16 = lane & 15;
  const int h = blockIdx.y, b = blockIdx.z;
  const int qbase = blockIdx.x * 128 + wave * 16;
  const size_t tok0 = (size_t)b * T;

  const bf16* Qp = qkv + tok0 * C3 + h * D;
  const bf16* Kp = qkv + tok0 * C3 + C + h * D;
  const bf16* Vp = qkv + tok0 * C3 + 2 * C + h * D;

  __shared__ bf16 Vt[8][32 * D];    // per-wave V tile: 32 keys x 64 d (4KB)
  __shared__ bf16 Ps[8][16 * 40];   // per-wave P tile: 16 rows x 32 keys
  bf16* Vtile = &Vt[wave][0];
  bf16* P = &Ps[wave][0];
#if USE_TDM
  const unsigned v_off = (unsigned)(uintptr_t)Vtile;
#endif

  v16bf qf[2];
#pragma unroll
  for (int t = 0; t < 2; ++t)
    qf[t] = load_frag(Qp + (size_t)(qbase + lane16) * C3 + t * 32, laneHi);

  v8f o[4] = {};
  float mrow[8], lrow[8];
#pragma unroll
  for (int j = 0; j < 8; ++j) { mrow[j] = -1e30f; lrow[j] = 0.0f; }

  const int nk = ((qbase + 16 + 31) >> 5) << 5;    // causal key bound
  for (int kb = 0; kb < nk; kb += 32) {
#if USE_TDM
    // kick V-tile DMA; overlaps with QK^T below
    tdm_load_2d_bf16(v_off, Vp + (size_t)kb * C3, D, 32,
                     (unsigned long long)C3, false, 0, 0);
#endif
    // ---- S = Q K^T (two 16-key tiles) ----
    v8f s[2] = {};
#pragma unroll
    for (int kt = 0; kt < 2; ++kt) {
      const bf16* krow = Kp + (size_t)(kb + kt * 16 + lane16) * C3;
      v16bf kf0 = load_frag(krow, laneHi);
      v16bf kf1 = load_frag(krow + 32, laneHi);
      s[kt] = wmma_bf16(qf[0], kf0, s[kt]);
      s[kt] = wmma_bf16(qf[1], kf1, s[kt]);
    }

    // ---- scale + causal mask ----
    const float scale = 0.125f;
    float sv[2][8];
    const bool need_mask = (kb + 31 > qbase);
#pragma unroll
    for (int kt = 0; kt < 2; ++kt)
#pragma unroll
      for (int j = 0; j < 8; ++j) {
        float v = s[kt][j] * scale;
        if (need_mask) {
          const int mg = qbase + laneHi * 8 + j;
          const int ng = kb + kt * 16 + lane16;
          if (ng > mg) v = -1e30f;
        }
        sv[kt][j] = v;
      }

    // ---- online softmax (row lives across a 16-lane half-wave) ----
#pragma unroll
    for (int j = 0; j < 8; ++j) {
      float mx = fmaxf(sv[0][j], sv[1][j]);
#pragma unroll
      for (int off = 1; off < 16; off <<= 1)
        mx = fmaxf(mx, __shfl_xor(mx, off, 16));
      const float newm = fmaxf(mrow[j], mx);
      const float p0 = __expf(sv[0][j] - newm);
      const float p1 = __expf(sv[1][j] - newm);
      sv[0][j] = p0; sv[1][j] = p1;
      float sum = p0 + p1;
#pragma unroll
      for (int off = 1; off < 16; off <<= 1)
        sum += __shfl_xor(sum, off, 16);
      const float alpha = __expf(mrow[j] - newm);
      lrow[j] = lrow[j] * alpha + sum;
      mrow[j] = newm;
#pragma unroll
      for (int t = 0; t < 4; ++t) o[t][j] *= alpha;
    }

    // ---- P -> LDS (C-layout out, A-layout back) ----
    __builtin_amdgcn_wave_barrier();
#pragma unroll
    for (int kt = 0; kt < 2; ++kt)
#pragma unroll
      for (int j = 0; j < 8; ++j)
        P[(laneHi * 8 + j) * 40 + kt * 16 + lane16] = f2bf(sv[kt][j]);
    __builtin_amdgcn_wave_barrier();
    v16bf pf = load_frag(P + lane16 * 40, laneHi);

#if USE_TDM
    __builtin_amdgcn_s_wait_tensorcnt(0);   // V tile landed in LDS
    asm volatile("" ::: "memory");
    __builtin_amdgcn_wave_barrier();
#endif

    // ---- O += P V : B fragment = V^T, read from LDS tile [key][d] ----
#pragma unroll
    for (int t = 0; t < 4; ++t) {
      v16bf vf;
#pragma unroll
      for (int i = 0; i < 8; ++i) {
        const int k0 = laneHi * 8 + i;
        const int k1 = 16 + laneHi * 8 + i;
#if USE_TDM
        vf[i]     = Vtile[k0 * D + t * 16 + lane16];
        vf[i + 8] = Vtile[k1 * D + t * 16 + lane16];
#else
        vf[i]     = Vp[(size_t)(kb + k0) * C3 + t * 16 + lane16];
        vf[i + 8] = Vp[(size_t)(kb + k1) * C3 + t * 16 + lane16];
#endif
      }
      o[t] = wmma_bf16(pf, vf, o[t]);
    }
#if USE_TDM
    __builtin_amdgcn_wave_barrier();        // V reads done before next DMA
#endif
  }

  // ---- epilogue: normalize, store y bf16 ----
#pragma unroll
  for (int t = 0; t < 4; ++t)
#pragma unroll
    for (int j = 0; j < 8; ++j) {
      const int row = qbase + laneHi * 8 + j;
      const float v = o[t][j] / lrow[j];
      y[(tok0 + row) * C + h * D + t * 16 + lane16] = f2bf(v);
    }
}

// ---------------------------------------------------------------------------
// Host launcher
// ---------------------------------------------------------------------------
extern "C" void kernel_launch(void* const* d_in, const int* in_sizes, int n_in,
                              void* d_out, int out_size, void* d_ws, size_t ws_size,
                              hipStream_t stream) {
  const float* x      = (const float*)d_in[0];   // [4,2048,1024]
  const float* w_attn = (const float*)d_in[1];   // [1024,3072]
  const float* w_proj = (const float*)d_in[2];   // [1024,1024]
  float* out = (float*)d_out;                    // [4,2048,1024]

  const int B = 4, T = 2048, C = 1024, H = 16;
  const size_t M = (size_t)B * T;                // 8192

  char* ws = (char*)d_ws;
  bf16* xb   = (bf16*)ws;  ws += M * C * sizeof(bf16);
  bf16* wab  = (bf16*)ws;  ws += (size_t)C * 3 * C * sizeof(bf16);
  bf16* wpb  = (bf16*)ws;  ws += (size_t)C * C * sizeof(bf16);
  bf16* qkvb = (bf16*)ws;  ws += M * 3 * C * sizeof(bf16);
  bf16* yb   = (bf16*)ws;

  auto cvt = [&](const float* src, bf16* dst, size_t n) {
    long long n8 = (long long)(n / 8);
    int blocks = (int)((n8 + 255) / 256);
    f32_to_bf16_vec<<<blocks, 256, 0, stream>>>(src, dst, n8);
  };
  cvt(x,      xb,  M * C);
  cvt(w_attn, wab, (size_t)C * 3 * C);
  cvt(w_proj, wpb, (size_t)C * C);

  gemm_bf16_wmma<true><<<dim3(3 * C / 128, (int)(M / 128)), 256, 0, stream>>>(
      xb, wab, qkvb, (int)M, 3 * C, C);

  flash_attn_wmma<<<dim3(T / 128, H, B), 256, 0, stream>>>(qkvb, yb);

  gemm_bf16_wmma<false><<<dim3(C / 128, (int)(M / 128)), 256, 0, stream>>>(
      yb, wpb, out, (int)M, C, C);
}
